// TERMEdgeMPNNLayer_64673617543294
// MI455X (gfx1250) — matrix operand order repaired and checked by reading, not compiled, not run on hardware
//
#include <hip/hip_runtime.h>
#include <hip/hip_bf16.h>
#include <math.h>

#define DEV __device__ __forceinline__

// Problem constants (match reference)
constexpr int Bb = 4, Tt = 48, Nn = 30, Kk = 30, Hh = 128, NINc = 384, FFc = 512;
constexpr int MTOT = Bb * Tt * Nn * Kk;   // 172800 edge rows
constexpr int MBLK = MTOT / 16;           // 10800 tiles of 16 rows
constexpr float EPSc = 1e-6f;

typedef __attribute__((ext_vector_type(16))) __bf16 v16bf;
typedef __attribute__((ext_vector_type(8)))  float  v8f;

union Frag {
  v16bf v;
  uint4 q[2];
};

DEV unsigned short f2bf(float f) {
  unsigned int u = __builtin_bit_cast(unsigned int, f);
  u += 0x7FFFu + ((u >> 16) & 1u);   // round-to-nearest-even
  return (unsigned short)(u >> 16);
}

// A-matrix fragment (16x32 bf16, CDNA5 layout) from LDS row-major [16][stride]
// lane l, elem e: m = l%16 ; k = ktile*32 + e + 8*((e>=8) + (l>=16))
DEV v16bf load_a_frag(const unsigned short* A, int stride, int kt, int lane) {
  int m  = lane & 15;
  int kb = kt * 32 + 8 * (lane >> 4);
  const unsigned short* p = A + m * stride + kb;
  Frag f;
  f.q[0] = *(const uint4*)(p);        // e = 0..7  -> k = kb + e
  f.q[1] = *(const uint4*)(p + 16);   // e = 8..15 -> k = kb + 16 + (e-8)
  return f.v;
}

// B-matrix fragment (32x16 bf16) from pre-packed weights: per (n0,kt) the 32
// lanes' 16 elems are contiguous (32B per lane -> 1KB coalesced per fragment)
DEV v16bf load_b_frag(const unsigned short* Wp, int nkt, int n0, int kt, int lane) {
  const unsigned short* p = Wp + ((((n0 * nkt + kt) << 5) + lane) << 4);
  Frag f;
  f.q[0] = *(const uint4*)(p);
  f.q[1] = *(const uint4*)(p + 8);
  return f.v;
}

DEV v8f wmma_bf16(v16bf a, v16bf b, v8f c) {
  return __builtin_amdgcn_wmma_f32_16x16x32_bf16(false, a, false, b, (short)0, c,
                                                 false, false);
}

// ---------------------------------------------------------------------------
// Prep: pack fp32 weight [K][N] -> bf16 fragments.
// Packed elem t = ((n0*nkt + kt)*32 + lane)*16 + e holds W[k][n] with
// k = kt*32 + (e&7) + 16*(e>=8) + 8*(lane>=16), n = n0*16 + (lane&15).
__global__ __launch_bounds__(256) void pack_w_kernel(const float* __restrict__ W,
                                                     unsigned short* __restrict__ dst,
                                                     int K, int N, int total) {
  int t = blockIdx.x * 256 + threadIdx.x;
  if (t >= total) return;
  int nkt  = K >> 5;
  int e    = t & 15;
  int lane = (t >> 4) & 31;
  int rem  = t >> 9;
  int kt   = rem % nkt;
  int n0   = rem / nkt;
  int k = kt * 32 + (e & 7) + 16 * (e >> 3) + 8 * (lane >> 4);
  int n = n0 * 16 + (lane & 15);
  dst[t] = f2bf(W[k * N + n]);
}

// Prep: inverse permutation. inv[bt][n][j] = k with E_idx[bt][n][k] == j.
__global__ __launch_bounds__(256) void inv_kernel(const int* __restrict__ E_idx,
                                                  int* __restrict__ inv, int total) {
  int t = blockIdx.x * 256 + threadIdx.x;
  if (t >= total) return;
  int k  = t % Kk;
  int n  = (t / Kk) % Nn;
  int bt = t / (Kk * Nn);
  int j  = E_idx[t];
  inv[(bt * Nn + n) * Nn + j] = k;
}

// ---------------------------------------------------------------------------
// Kernel 1: dh = W3(relu(W2(relu(W1(h_EV))))). 16 rows x 128 cols per block.
__global__ __launch_bounds__(256) void mlp_kernel(
    const float* __restrict__ hEV,
    const unsigned short* __restrict__ W1p, const float* __restrict__ W1b,
    const unsigned short* __restrict__ W2p, const float* __restrict__ W2b,
    const unsigned short* __restrict__ W3p, const float* __restrict__ W3b,
    float* __restrict__ dh) {
  __shared__ __align__(16) unsigned short A1[16 * NINc];
  __shared__ __align__(16) unsigned short A2[16 * Hh];
  __shared__ __align__(16) unsigned short A3[16 * Hh];

  const int row0 = blockIdx.x * 16;
  const int tid  = threadIdx.x;
  const int lane = tid & 31;
  const int w    = tid >> 5;

  // Stage 16x384 activation tile as bf16 in LDS
  for (int i = tid; i < 16 * NINc; i += 256)
    A1[i] = f2bf(hEV[(size_t)row0 * NINc + i]);
  __syncthreads();

  const int ncol = w * 16 + (lane & 15);
  const int mrow = 8 * (lane >> 4);

  // GEMM1: [16x384] x [384x16]  (12 WMMAs)
  v8f c = {};
  for (int kt = 0; kt < NINc / 32; ++kt)
    c = wmma_bf16(load_a_frag(A1, NINc, kt, lane),
                  load_b_frag(W1p, NINc / 32, w, kt, lane), c);
  {
    float bias = W1b[ncol];
    for (int r = 0; r < 8; ++r)
      A2[(r + mrow) * Hh + ncol] = f2bf(fmaxf(c[r] + bias, 0.0f));
  }
  __syncthreads();

  // GEMM2: [16x128] x [128x16]  (4 WMMAs)
  c = (v8f){};
  for (int kt = 0; kt < Hh / 32; ++kt)
    c = wmma_bf16(load_a_frag(A2, Hh, kt, lane),
                  load_b_frag(W2p, Hh / 32, w, kt, lane), c);
  {
    float bias = W2b[ncol];
    for (int r = 0; r < 8; ++r)
      A3[(r + mrow) * Hh + ncol] = f2bf(fmaxf(c[r] + bias, 0.0f));
  }
  __syncthreads();

  // GEMM3: [16x128] x [128x16]  (4 WMMAs) -> dh (fp32, global)
  c = (v8f){};
  for (int kt = 0; kt < Hh / 32; ++kt)
    c = wmma_bf16(load_a_frag(A3, Hh, kt, lane),
                  load_b_frag(W3p, Hh / 32, w, kt, lane), c);
  {
    float bias = W3b[ncol];
    for (int r = 0; r < 8; ++r)
      dh[(size_t)(row0 + r + mrow) * Hh + ncol] = c[r] + bias;
  }
}

// ---------------------------------------------------------------------------
// Kernel 2: reverse-edge merge + mask_attend + residual + LayerNorm -> h (d_out)
__global__ __launch_bounds__(128) void merge_norm_kernel(
    const float* __restrict__ hE, const float* __restrict__ dh,
    const int* __restrict__ E_idx, const int* __restrict__ inv,
    const unsigned char* __restrict__ mask_attend,
    const float* __restrict__ g0, const float* __restrict__ b0,
    float* __restrict__ out) {
  const int row = blockIdx.x;
  const int h   = threadIdx.x;

  const int i  = (row / Kk) % Nn;
  const int bt = row / (Kk * Nn);
  const int j  = E_idx[row];
  const int kp = inv[(bt * Nn + j) * Nn + i];        // E_idx[bt][j][kp] == i
  const size_t rrow = (size_t)((bt * Nn + j) * Kk + kp);

  float d  = dh[(size_t)row * Hh + h];
  float rv = dh[rrow * Hh + h];
  float m  = (rv != 0.0f) ? (d + rv) * 0.5f : d;     // elementwise, as reference
  if (!mask_attend[row]) m = 0.0f;
  float x = hE[(size_t)row * Hh + h] + m;

  // reduction over 128 lanes (4 waves)
  float s = x, s2 = x * x;
  for (int o = 16; o > 0; o >>= 1) {
    s  += __shfl_down(s, o);
    s2 += __shfl_down(s2, o);
  }
  __shared__ float ps[4], ps2[4];
  __shared__ float mu_s, rs_s;
  if ((h & 31) == 0) { ps[h >> 5] = s; ps2[h >> 5] = s2; }
  __syncthreads();
  if (h == 0) {
    float S = ps[0] + ps[1] + ps[2] + ps[3];
    float S2 = ps2[0] + ps2[1] + ps2[2] + ps2[3];
    float mu = S / (float)Hh;
    float var = fmaxf((S2 - (float)Hh * mu * mu) / (float)(Hh - 1), 0.0f); // ddof=1
    mu_s = mu;
    rs_s = 1.0f / (sqrtf(var) + EPSc);
  }
  __syncthreads();
  out[(size_t)row * Hh + h] = g0[h] * (x - mu_s) * rs_s + b0[h];
}

// ---------------------------------------------------------------------------
// Kernel 3: FFN (128->512->128) + residual + LayerNorm + mask_E, in place on d_out
__global__ __launch_bounds__(256) void ffn_norm_kernel(
    float* __restrict__ io,
    const unsigned short* __restrict__ Winp, const float* __restrict__ bin,
    const unsigned short* __restrict__ Woutp, const float* __restrict__ bout,
    const float* __restrict__ g1, const float* __restrict__ b1,
    const unsigned char* __restrict__ maskE) {
  __shared__ __align__(16) unsigned short hb[16 * Hh];   // bf16 activations
  __shared__ __align__(16) float          hf[16 * Hh];   // fp32 residual copy
  __shared__ __align__(16) unsigned short A2[16 * FFc];  // hidden (bf16)
  __shared__ __align__(16) float          xb[16 * Hh];   // pre-norm x
  __shared__ float red[16 * 16 * 2];
  __shared__ float rowmu[16], rowrs[16];

  const int row0 = blockIdx.x * 16;
  const int tid  = threadIdx.x;
  const int lane = tid & 31;
  const int w    = tid >> 5;
  const int mrow = 8 * (lane >> 4);

  for (int i = tid; i < 16 * Hh; i += 256) {
    float v = io[(size_t)row0 * Hh + i];
    hf[i] = v;
    hb[i] = f2bf(v);
  }
  __syncthreads();

  // FFN-in: [16x128] x [128x512] ; each wave owns 4 n-tiles (16 WMMAs/wave)
  for (int t = 0; t < 4; ++t) {
    int n0 = w * 4 + t;
    v8f c = {};
    for (int kt = 0; kt < Hh / 32; ++kt)
      c = wmma_bf16(load_a_frag(hb, Hh, kt, lane),
                    load_b_frag(Winp, Hh / 32, n0, kt, lane), c);
    int nc = n0 * 16 + (lane & 15);
    float bias = bin[nc];
    for (int r = 0; r < 8; ++r)
      A2[(r + mrow) * FFc + nc] = f2bf(fmaxf(c[r] + bias, 0.0f));
  }
  __syncthreads();

  // FFN-out: [16x512] x [512x16] per wave (16 WMMAs) + bias + residual
  {
    v8f c = {};
    for (int kt = 0; kt < FFc / 32; ++kt)
      c = wmma_bf16(load_a_frag(A2, FFc, kt, lane),
                    load_b_frag(Woutp, FFc / 32, w, kt, lane), c);
    int nc = w * 16 + (lane & 15);
    float bias = bout[nc];
    for (int r = 0; r < 8; ++r)
      xb[(r + mrow) * Hh + nc] = c[r] + bias + hf[(r + mrow) * Hh + nc];
  }
  __syncthreads();

  // LayerNorm (ddof=1) per row; 16 threads per row, 8 cols each
  const int rr  = tid >> 4;
  const int sub = tid & 15;
  {
    float s = 0.f, s2 = 0.f;
    for (int cidx = sub * 8; cidx < sub * 8 + 8; ++cidx) {
      float v = xb[rr * Hh + cidx];
      s += v; s2 += v * v;
    }
    red[(rr * 16 + sub) * 2]     = s;
    red[(rr * 16 + sub) * 2 + 1] = s2;
  }
  __syncthreads();
  if (sub == 0) {
    float S = 0.f, S2 = 0.f;
    for (int q = 0; q < 16; ++q) {
      S  += red[(rr * 16 + q) * 2];
      S2 += red[(rr * 16 + q) * 2 + 1];
    }
    float mu = S / (float)Hh;
    float var = fmaxf((S2 - (float)Hh * mu * mu) / (float)(Hh - 1), 0.0f);
    rowmu[rr] = mu;
    rowrs[rr] = 1.0f / (sqrtf(var) + EPSc);
  }
  __syncthreads();
  {
    int grow = row0 + rr;
    float mE = maskE[grow / Kk] ? 1.0f : 0.0f;
    float mu = rowmu[rr], rs = rowrs[rr];
    for (int cidx = sub * 8; cidx < sub * 8 + 8; ++cidx) {
      float v = g1[cidx] * (xb[rr * Hh + cidx] - mu) * rs + b1[cidx];
      io[(size_t)grow * Hh + cidx] = mE * v;
    }
  }
}

// ---------------------------------------------------------------------------
static inline size_t align256(size_t x) { return (x + 255) & ~(size_t)255; }

extern "C" void kernel_launch(void* const* d_in, const int* in_sizes, int n_in,
                              void* d_out, int out_size, void* d_ws, size_t ws_size,
                              hipStream_t stream) {
  (void)in_sizes; (void)n_in; (void)out_size; (void)ws_size;

  const float* hE     = (const float*)d_in[0];
  const float* hEV    = (const float*)d_in[1];
  const int*   E_idx  = (const int*)d_in[2];
  const unsigned char* maskE  = (const unsigned char*)d_in[3];
  const unsigned char* maskA  = (const unsigned char*)d_in[4];
  const float* W1w = (const float*)d_in[5];
  const float* W1b = (const float*)d_in[6];
  const float* W2w = (const float*)d_in[7];
  const float* W2b = (const float*)d_in[8];
  const float* W3w = (const float*)d_in[9];
  const float* W3b = (const float*)d_in[10];
  const float* Win  = (const float*)d_in[11];
  const float* bin  = (const float*)d_in[12];
  const float* Wout = (const float*)d_in[13];
  const float* bout = (const float*)d_in[14];
  const float* g0 = (const float*)d_in[15];
  const float* b0 = (const float*)d_in[16];
  const float* g1 = (const float*)d_in[17];
  const float* b1 = (const float*)d_in[18];
  float* out = (float*)d_out;

  // Workspace carve-up
  char* ws = (char*)d_ws;
  size_t off = 0;
  float* dh = (float*)(ws + off);            off = align256(off + (size_t)MTOT * Hh * 4);
  unsigned short* W1p  = (unsigned short*)(ws + off); off = align256(off + (size_t)NINc * Hh * 2);
  unsigned short* W2p  = (unsigned short*)(ws + off); off = align256(off + (size_t)Hh * Hh * 2);
  unsigned short* W3p  = (unsigned short*)(ws + off); off = align256(off + (size_t)Hh * Hh * 2);
  unsigned short* Winp = (unsigned short*)(ws + off); off = align256(off + (size_t)Hh * FFc * 2);
  unsigned short* Woutp= (unsigned short*)(ws + off); off = align256(off + (size_t)FFc * Hh * 2);
  int* inv = (int*)(ws + off);               off = align256(off + (size_t)Bb * Tt * Nn * Nn * 4);

  // Prep: weight repack (bf16 WMMA-B fragment order) + inverse permutation
  auto packs = [&](const float* W, unsigned short* dst, int K, int N) {
    int total = K * N;
    pack_w_kernel<<<(total + 255) / 256, 256, 0, stream>>>(W, dst, K, N, total);
  };
  packs(W1w, W1p, NINc, Hh);
  packs(W2w, W2p, Hh, Hh);
  packs(W3w, W3p, Hh, Hh);
  packs(Win, Winp, Hh, FFc);
  packs(Wout, Woutp, FFc, Hh);
  inv_kernel<<<(MTOT + 255) / 256, 256, 0, stream>>>(E_idx, inv, MTOT);

  // Stage 1: edge-message MLP (bf16 WMMA, fp32 accum)
  mlp_kernel<<<MBLK, 256, 0, stream>>>(hEV, W1p, W1b, W2p, W2b, W3p, W3b, dh);

  // Stage 2: reverse-edge merge + mask + residual + LayerNorm -> d_out
  merge_norm_kernel<<<MTOT, 128, 0, stream>>>(hE, dh, E_idx, inv, maskA, g0, b0, out);

  // Stage 3: FFN + residual + LayerNorm + node mask, in place on d_out
  ffn_norm_kernel<<<MBLK, 256, 0, stream>>>(out, Winp, bin, Woutp, bout, g1, b1, maskE);
}
